// Decoder_75419625718018
// MI455X (gfx1250) — compile-verified, matrix operand order checked by reading, not run on hardware
//
#include <hip/hip_runtime.h>

// ---------------- types ----------------
typedef __attribute__((ext_vector_type(16))) __bf16 v16bf;
typedef __attribute__((ext_vector_type(8)))  __bf16 v8bf;
typedef __attribute__((ext_vector_type(8)))  float  v8f;
typedef __attribute__((ext_vector_type(4)))  int    v4i;

// dims
#define NB   64      // batch
#define TT   512     // encoder time
#define LL   256     // decoder steps
#define KS   256     // key size / hidden per cell
#define HID  512     // embedding dim
#define GATES 1024   // 4*KS
#define VOC  10000

#if __has_builtin(__builtin_amdgcn_global_load_async_to_lds_b128) && \
    __has_builtin(__builtin_amdgcn_s_wait_asynccnt)
#define HAVE_ASYNC_LDS 1
#else
#define HAVE_ASYNC_LDS 0
#endif

__device__ __forceinline__ unsigned short f2bf(float f) {
    union { float f; unsigned u; } v; v.f = f;
    unsigned r = v.u + 0x7FFFu + ((v.u >> 16) & 1u);   // round-to-nearest-even
    return (unsigned short)(r >> 16);
}

// Load a 16x32 (MxK) bf16 fragment in the CDNA5 WMMA A/B register layout.
// Row index (M for A, N for B) = row0 + (lane&15); K chunks per ISA:
// lanes 0-15 hold K = {0..7, 16..23}, lanes 16-31 hold K = {8..15, 24..31}.
__device__ __forceinline__ v16bf frag_ld(const unsigned short* base, int row0, int ld, int k0) {
    int lane = threadIdx.x & 31;
    const unsigned short* q = base + (size_t)(row0 + (lane & 15)) * (size_t)ld
                                   + k0 + ((lane >> 4) << 3);
    v8bf lo = *(const v8bf*)q;
    v8bf hi = *(const v8bf*)(q + 16);
    v16bf r;
#pragma unroll
    for (int i = 0; i < 8; ++i) { r[i] = lo[i]; r[i + 8] = hi[i]; }
    return r;
}

__device__ __forceinline__ v8f wmma_bf16(v16bf a, v16bf b, v8f c) {
    return __builtin_amdgcn_wmma_f32_16x16x32_bf16(false, a, false, b, (short)0, c, false, false);
}

// ---------------- prep kernels ----------------
__global__ void k_cvt_bf16(unsigned short* __restrict__ dst, const float* __restrict__ src, int n) {
    int i = blockIdx.x * blockDim.x + threadIdx.x;
    if (i < n) dst[i] = f2bf(src[i]);
}

// W1bf[j, 0:768] = W_ih1[j,:], W1bf[j, 768:1024] = W_hh1[j,:]
__global__ void k_pack_w1(unsigned short* __restrict__ w, const float* __restrict__ wih,
                          const float* __restrict__ whh) {
    int i = blockIdx.x * blockDim.x + threadIdx.x;   // 1024*1024
    int j = i >> 10, k = i & 1023;
    float v = (k < 768) ? wih[j * 768 + k] : whh[j * 256 + (k - 768)];
    w[i] = f2bf(v);
}

// W2bf[j, 0:256] = W_ih2[j,:], W2bf[j, 256:512] = W_hh2[j,:]
__global__ void k_pack_w2(unsigned short* __restrict__ w, const float* __restrict__ wih,
                          const float* __restrict__ whh) {
    int i = blockIdx.x * blockDim.x + threadIdx.x;   // 1024*512
    int j = i >> 9, k = i & 511;
    float v = (k < 256) ? wih[j * 256 + k] : whh[j * 256 + (k - 256)];
    w[i] = f2bf(v);
}

// embseq[l,n,k] = bf16(E[tok, k]); tok = 33 (SOS) at l=0, else text[n, l-1]
__global__ void k_embseq(unsigned short* __restrict__ dst, const float* __restrict__ E,
                         const int* __restrict__ text) {
    size_t i = (size_t)blockIdx.x * blockDim.x + threadIdx.x;  // 256*64*512
    int k = (int)(i & 511);
    int n = (int)((i >> 9) & 63);
    int l = (int)(i >> 15);
    int tok = (l == 0) ? 33 : text[n * LL + (l - 1)];
    dst[i] = f2bf(E[(size_t)tok * HID + k]);
}

// ctx0 = mean over valid t of V[n,t,:]; zero recurrent state.
__global__ void k_init(const float* __restrict__ Vmat, const int* __restrict__ lens,
                       unsigned short* __restrict__ ctxbf, unsigned short* __restrict__ h1bf,
                       unsigned short* __restrict__ h2bf, float* __restrict__ c1,
                       float* __restrict__ c2) {
    int n = blockIdx.x, v = threadIdx.x;
    int len = lens[n];
    const float* vb = Vmat + (size_t)n * TT * KS + v;
    float acc = 0.f;
    for (int t = 0; t < len; ++t) acc += vb[(size_t)t * KS];
    acc /= (float)len;
    int idx = n * KS + v;
    ctxbf[idx] = f2bf(acc);
    h1bf[idx] = 0; h2bf[idx] = 0;
    c1[idx] = 0.f; c2[idx] = 0.f;
}

// ---------------- fused LSTM cell: WMMA gate GEMM + activation ----------------
// gates = concat_segments(A) @ W^T (bf16 -> f32 WMMA), accumulators parked in
// LDS, then i/f/g/o activation applied in-block. grid = 16 blocks:
// (bx&3) -> 16-row n-tile, (bx>>2) -> 64-wide k-slice replicated across the
// 4 gate quadrants. 256 threads = 8 waves; wave w: quadrant w>>1, col-pair w&1.
__global__ void k_lstm_fused(const unsigned short* __restrict__ a0, int l0,
                             const unsigned short* __restrict__ a1, int l1,
                             const unsigned short* __restrict__ a2, int l2,
                             const unsigned short* __restrict__ W, int ldw,
                             const float* __restrict__ bih, const float* __restrict__ bhh,
                             float* __restrict__ c, float* __restrict__ hf,
                             unsigned short* __restrict__ d0, int ld0,
                             unsigned short* __restrict__ d1, int ld1) {
    __shared__ float gl[4 * 16 * 64];   // [quadrant][row][col] = 16 KB
    int bx = blockIdx.x;
    int n0 = (bx & 3) * 16;
    int kk0 = (bx >> 2) * 64;
    int wave = threadIdx.x >> 5;
    int q = wave >> 1;
    int cp = wave & 1;
    int j0 = q * 256 + kk0 + cp * 32;

    v8f acc0 = {0.f,0.f,0.f,0.f,0.f,0.f,0.f,0.f};
    v8f acc1 = acc0;
    const unsigned short* segs[3] = {a0, a1, a2};
    int slen[3] = {l0, l1, l2};
    int kbase = 0;
    for (int s = 0; s < 3; ++s) {
        const unsigned short* A = segs[s];
        int L = slen[s];
        if (!A || L == 0) continue;
        for (int k = 0; k < L; k += 32) {
            v16bf af = frag_ld(A, n0, L, k);
            v16bf b0 = frag_ld(W, j0,      ldw, kbase + k);
            v16bf b1 = frag_ld(W, j0 + 16, ldw, kbase + k);
            acc0 = wmma_bf16(af, b0, acc0);
            acc1 = wmma_bf16(af, b1, acc1);
        }
        kbase += L;
    }
    {   // park accumulators in LDS: gl[q][m][cp*32 + tile*16 + j]
        int lane = threadIdx.x & 31;
        int j = lane & 15;
        int mr = (lane >> 4) * 8;
#pragma unroll
        for (int i = 0; i < 8; ++i) {
            int m = mr + i;
            gl[q * 1024 + m * 64 + cp * 32 + j]      = acc0[i];
            gl[q * 1024 + m * 64 + cp * 32 + 16 + j] = acc1[i];
        }
    }
    __syncthreads();
    // activation: 16 rows x 64 k-slice = 1024 elems, 4 per thread
#pragma unroll
    for (int e = 0; e < 4; ++e) {
        int idx = threadIdx.x + e * 256;
        int m = idx >> 6;
        int kl = idx & 63;
        int kg = kk0 + kl;
        float gi = gl[          m * 64 + kl] + bih[kg]       + bhh[kg];
        float gf = gl[1 * 1024 + m * 64 + kl] + bih[256 + kg] + bhh[256 + kg];
        float gg = gl[2 * 1024 + m * 64 + kl] + bih[512 + kg] + bhh[512 + kg];
        float go = gl[3 * 1024 + m * 64 + kl] + bih[768 + kg] + bhh[768 + kg];
        float si = 1.f / (1.f + __expf(-gi));
        float sf = 1.f / (1.f + __expf(-gf));
        float so = 1.f / (1.f + __expf(-go));
        int n = n0 + m;
        int cidx = n * KS + kg;
        float cn = sf * c[cidx] + si * tanhf(gg);
        float hn = so * tanhf(cn);
        c[cidx] = cn;
        if (hf) hf[cidx] = hn;
        unsigned short hb = f2bf(hn);
        d0[(size_t)n * ld0 + kg] = hb;
        if (d1) d1[(size_t)n * ld1 + kg] = hb;
    }
}

// ---------------- attention (one block per batch row) ----------------
__global__ void k_attend(const float* __restrict__ h2, const float* __restrict__ Kmat,
                         const float* __restrict__ Vmat, const int* __restrict__ lens,
                         unsigned short* __restrict__ ctxbf, unsigned short* __restrict__ rec) {
    int n = blockIdx.x, tid = threadIdx.x;
    __shared__ float q[KS];
    __shared__ float attn[TT];
    __shared__ float red[256];
    q[tid] = h2[n * KS + tid];
    __syncthreads();
    int len = lens[n];
    float e[2];
#pragma unroll
    for (int r = 0; r < 2; ++r) {
        int t = tid + r * 256;
        const float* kr = Kmat + ((size_t)n * TT + t) * KS;
        float acc = 0.f;
        for (int k = 0; k < KS; k += 4)
            acc += kr[k] * q[k] + kr[k+1] * q[k+1] + kr[k+2] * q[k+2] + kr[k+3] * q[k+3];
        e[r] = (t < len) ? acc : -1e30f;
    }
    red[tid] = fmaxf(e[0], e[1]);
    __syncthreads();
    for (int s = 128; s > 0; s >>= 1) {
        if (tid < s) red[tid] = fmaxf(red[tid], red[tid + s]);
        __syncthreads();
    }
    float mx = red[0];
    __syncthreads();
    float ex0 = __expf(e[0] - mx), ex1 = __expf(e[1] - mx);
    red[tid] = ex0 + ex1;
    __syncthreads();
    for (int s = 128; s > 0; s >>= 1) {
        if (tid < s) red[tid] += red[tid + s];
        __syncthreads();
    }
    float inv = 1.f / red[0];
    attn[tid] = ex0 * inv;
    attn[tid + 256] = ex1 * inv;
    __syncthreads();
    const float* vb = Vmat + (size_t)n * TT * KS + tid;
    float acc = 0.f;
    for (int t = 0; t < TT; ++t) acc += attn[t] * vb[(size_t)t * KS];
    unsigned short b = f2bf(acc);
    ctxbf[n * KS + tid] = b;
    rec[(size_t)n * HID + tid] = b;
}

// ---------------- logits GEMM: [16384,512] x [512,10000] ----------------
// block = 256 thr (8 waves), tile 64(M) x 128(V), LDS-staged k-chunks of 32.
// A tile staged with GLOBAL_LOAD_ASYNC_TO_LDS_B128 (ASYNCcnt) when available.
__global__ void k_logits(const unsigned short* __restrict__ A,   // h2ctx [L*N, 512]
                         const unsigned short* __restrict__ B,   // Ebf  [VOC, 512]
                         const float* __restrict__ bias, float* __restrict__ out) {
    __shared__ __align__(16) unsigned short As[64 * 32];
    __shared__ __align__(16) unsigned short Bs[128 * 32];
    int m0 = blockIdx.x * 64;
    int v0 = blockIdx.y * 128;
    int wave = threadIdx.x >> 5;
    int wm = wave >> 2;     // 0..1
    int wn = wave & 3;      // 0..3
    v8f acc[2][2];
#pragma unroll
    for (int i = 0; i < 2; ++i)
#pragma unroll
        for (int j = 0; j < 2; ++j) acc[i][j] = (v8f){0.f,0.f,0.f,0.f,0.f,0.f,0.f,0.f};

    for (int k0 = 0; k0 < HID; k0 += 32) {
        __syncthreads();
        {   // stage A tile 64x32 (one 16B chunk per thread)
            int row = threadIdx.x >> 2;
            int col = (threadIdx.x & 3) * 8;
            const unsigned short* gsrc = A + (size_t)(m0 + row) * HID + k0 + col;
            unsigned short* ldst = As + row * 32 + col;
#if HAVE_ASYNC_LDS
            // signature: (v4i addrspace(1)*, v4i addrspace(3)*, imm offset, imm cpol)
            __attribute__((address_space(1))) v4i* g1 =
                (__attribute__((address_space(1))) v4i*)
                (const __attribute__((address_space(1))) void*)gsrc;
            __attribute__((address_space(3))) v4i* l3 =
                (__attribute__((address_space(3))) v4i*)
                (__attribute__((address_space(3))) void*)ldst;
            __builtin_amdgcn_global_load_async_to_lds_b128(g1, l3, 0, 0);
#else
            *(uint4*)ldst = *(const uint4*)gsrc;
#endif
        }
#pragma unroll
        for (int r = 0; r < 2; ++r) {   // stage B tile 128x32 (guard V edge)
            int t = threadIdx.x + r * 256;
            int row = t >> 2;
            int col = (t & 3) * 8;
            int v = v0 + row;
            uint4 val = {0u, 0u, 0u, 0u};
            if (v < VOC) val = *(const uint4*)(B + (size_t)v * HID + k0 + col);
            *(uint4*)(Bs + row * 32 + col) = val;
        }
#if HAVE_ASYNC_LDS
        __builtin_amdgcn_s_wait_asynccnt(0);
#endif
        __syncthreads();
        v16bf a0 = frag_ld(As, wm * 32,      32, 0);
        v16bf a1 = frag_ld(As, wm * 32 + 16, 32, 0);
        v16bf b0 = frag_ld(Bs, wn * 32,      32, 0);
        v16bf b1 = frag_ld(Bs, wn * 32 + 16, 32, 0);
        acc[0][0] = wmma_bf16(a0, b0, acc[0][0]);
        acc[0][1] = wmma_bf16(a0, b1, acc[0][1]);
        acc[1][0] = wmma_bf16(a1, b0, acc[1][0]);
        acc[1][1] = wmma_bf16(a1, b1, acc[1][1]);
    }
    int lane = threadIdx.x & 31;
    int j = lane & 15;
    int mr = (lane >> 4) * 8;
#pragma unroll
    for (int i = 0; i < 2; ++i)
#pragma unroll
        for (int jj = 0; jj < 2; ++jj)
#pragma unroll
            for (int e = 0; e < 8; ++e) {
                int m = m0 + wm * 32 + i * 16 + mr + e;   // row = l*64 + n
                int v = v0 + wn * 32 + jj * 16 + j;
                if (v < VOC) {
                    int n = m & 63, l = m >> 6;
                    out[((size_t)n * LL + l) * VOC + v] = acc[i][jj][e] + bias[v];
                }
            }
}

// ---------------- host ----------------
extern "C" void kernel_launch(void* const* d_in, const int* in_sizes, int n_in,
                              void* d_out, int out_size, void* d_ws, size_t ws_size,
                              hipStream_t stream) {
    (void)in_sizes; (void)n_in; (void)out_size; (void)ws_size;
    const float* Vmat  = (const float*)d_in[0];
    const float* Kmat  = (const float*)d_in[1];
    const int*   lens  = (const int*)d_in[2];
    const int*   text  = (const int*)d_in[3];
    const float* E     = (const float*)d_in[4];
    const float* b_out = (const float*)d_in[5];
    const float* W_ih1 = (const float*)d_in[6];
    const float* W_hh1 = (const float*)d_in[7];
    const float* b_ih1 = (const float*)d_in[8];
    const float* b_hh1 = (const float*)d_in[9];
    const float* W_ih2 = (const float*)d_in[10];
    const float* W_hh2 = (const float*)d_in[11];
    const float* b_ih2 = (const float*)d_in[12];
    const float* b_hh2 = (const float*)d_in[13];
    float* out = (float*)d_out;

    char* ws = (char*)d_ws;
    size_t off = 0;
    auto alloc = [&](size_t bytes) -> void* {
        void* p = ws + off;
        off = (off + bytes + 255) & ~(size_t)255;
        return p;
    };
    unsigned short* Ebf    = (unsigned short*)alloc((size_t)VOC * HID * 2);
    unsigned short* W1bf   = (unsigned short*)alloc((size_t)GATES * GATES * 2);
    unsigned short* W2bf   = (unsigned short*)alloc((size_t)GATES * HID * 2);
    unsigned short* embseq = (unsigned short*)alloc((size_t)LL * NB * HID * 2);
    unsigned short* h2ctx  = (unsigned short*)alloc((size_t)LL * NB * HID * 2);
    unsigned short* ctxbf  = (unsigned short*)alloc((size_t)NB * KS * 2);
    unsigned short* h1p[2] = {(unsigned short*)alloc((size_t)NB * KS * 2),
                              (unsigned short*)alloc((size_t)NB * KS * 2)};
    unsigned short* h2p[2] = {(unsigned short*)alloc((size_t)NB * KS * 2),
                              (unsigned short*)alloc((size_t)NB * KS * 2)};
    float*          c1     = (float*)alloc((size_t)NB * KS * 4);
    float*          c2     = (float*)alloc((size_t)NB * KS * 4);
    float*          h2f    = (float*)alloc((size_t)NB * KS * 4);

    // ---- prep ----
    k_cvt_bf16<<<(VOC * HID + 255) / 256, 256, 0, stream>>>(Ebf, E, VOC * HID);
    k_pack_w1<<<(GATES * GATES) / 256, 256, 0, stream>>>(W1bf, W_ih1, W_hh1);
    k_pack_w2<<<(GATES * HID) / 256, 256, 0, stream>>>(W2bf, W_ih2, W_hh2);
    k_embseq<<<((size_t)LL * NB * HID) / 256, 256, 0, stream>>>(embseq, E, text);
    k_init<<<NB, KS, 0, stream>>>(Vmat, lens, ctxbf, h1p[0], h2p[0], c1, c2);

    // ---- sequential recurrence (3 launches/step) ----
    for (int t = 0; t < LL; ++t) {
        const unsigned short* emb_t = embseq + (size_t)t * NB * HID;
        unsigned short* rec_t = h2ctx + (size_t)t * NB * HID;
        unsigned short* h1in = h1p[t & 1],  *h1out = h1p[(t & 1) ^ 1];
        unsigned short* h2in = h2p[t & 1],  *h2out = h2p[(t & 1) ^ 1];
        // cell 1: A = [emb(512) | ctx(256) | h1(256)], W1 [1024,1024]
        k_lstm_fused<<<16, 256, 0, stream>>>(emb_t, HID, ctxbf, KS, h1in, KS,
                                             W1bf, GATES, b_ih1, b_hh1, c1,
                                             (float*)nullptr, h1out, KS,
                                             (unsigned short*)nullptr, 0);
        // cell 2: A = [h1(256) | h2(256)], W2 [1024,512]; records h2 -> h2ctx
        k_lstm_fused<<<16, 256, 0, stream>>>(h1out, KS, h2in, KS,
                                             (const unsigned short*)nullptr, 0,
                                             W2bf, HID, b_ih2, b_hh2, c2,
                                             h2f, h2out, KS, rec_t, HID);
        // attention: writes ctxbf (next step input) + rec_t[:,256:512]
        k_attend<<<NB, 256, 0, stream>>>(h2f, Kmat, Vmat, lens, ctxbf, rec_t + KS);
    }

    // ---- batched output projection (tied embedding), 168 GFLOP bf16 WMMA ----
    dim3 lgrid((LL * NB) / 64, (VOC + 127) / 128);
    k_logits<<<lgrid, 256, 0, stream>>>(h2ctx, Ebf, b_out, out);
}
